// SelfAttentionV3_7164005449950
// MI455X (gfx1250) — matrix operand order repaired and checked by reading, not compiled, not run on hardware
//
#include <hip/hip_runtime.h>

// ---------------------------------------------------------------------------
// Self-attention forward for B=4, S=4096, D=512 on gfx1250 (MI455X).
// fp32 -> bf16 WMMA (v_wmma_f32_16x16x32_bf16) with fp32 accumulation,
// flash-attention streaming (score matrix never hits HBM), and a software-
// pipelined Tensor Data Mover: the next K tile's DMA overlaps V staging and
// the PV WMMA phase (separate 32KB K and V^T LDS buffers, 72KB total).
// ---------------------------------------------------------------------------

#define S_LEN   4096
#define D_HID   512
#define N_QKV   1536
#define SCALE_F 0.044194173824159216f   /* 1/sqrt(512) */
#define MASKFILL (-1e20f)

typedef __attribute__((ext_vector_type(16))) __bf16        v16bf;
typedef __attribute__((ext_vector_type(8)))  float         v8f;
typedef __attribute__((ext_vector_type(4)))  unsigned int  uv4;
typedef __attribute__((ext_vector_type(4)))  float         fv4;
typedef __attribute__((ext_vector_type(8)))  int           iv8;
typedef __attribute__((ext_vector_type(4)))  int           iv4;

#if __has_builtin(__builtin_amdgcn_tensor_load_to_lds) && \
    __has_builtin(__builtin_amdgcn_s_wait_tensorcnt)
#define USE_TDM 1
#endif

union Frag {             // one 16x32 bf16 A-fragment or 32x16 bf16 B-fragment
    v16bf v;
    uv4   q[2];
};

union U4H {              // 16-byte chunk viewed as 8 bf16 halves
    uv4            q;
    unsigned short h[8];
};

__device__ __forceinline__ unsigned short f2bf(float x) {
    union { float f; unsigned int u; } v; v.f = x;
    unsigned int r = v.u + 0x7FFFu + ((v.u >> 16) & 1u);   // round-nearest-even
    return (unsigned short)(r >> 16);
}

// reuse hints must be immediates -> template parameters
template <bool RA, bool RB>
__device__ __forceinline__ v8f wmma_bf16(const Frag& a, const Frag& b, v8f c) {
    return __builtin_amdgcn_wmma_f32_16x16x32_bf16(
        false, a.v, false, b.v, (short)0, c, RA, RB);
}

#ifdef USE_TDM
// One 2D TDM descriptor: 512x32 bf16 tile, tensor row stride 1536 elements.
__device__ __forceinline__ void tdm_load_tile(const unsigned short* gsrc,
                                              const unsigned short* ldst) {
    unsigned long long ga = (unsigned long long)(const void*)gsrc;
    unsigned ldsa = (unsigned)(unsigned long long)(const void*)ldst;
    uv4 g0 = {
        1u,                                     // count=1 (valid D#)
        ldsa,                                   // lds_addr (bytes)
        (unsigned)(ga & 0xffffffffu),           // global_addr[31:0]
        (unsigned)((ga >> 32) & 0x01ffffffu)    // global_addr[56:32]
            | (2u << 30)                        // type=2 ("image")
    };
    iv8 g1 = {
        (int)0x00010000,        // data_size=1 -> 2-byte elements
        (int)(512u << 16),      // tensor_dim0[15:0]=512 @bits[63:48]
        (int)(32u  << 16),      // tensor_dim1[15:0]=32  @bits[95:80]
        (int)(512u << 16),      // tile_dim0=512 @bits[127:112]
        32,                     // tile_dim1=32
        1536,                   // tensor_dim0_stride[31:0]=1536 elems
        0, 0
    };
    iv4 gz = { 0, 0, 0, 0 };    // 2D tensor: groups 2/3 unused
#if defined(__clang_major__) && (__clang_major__ >= 23)
    iv8 gz8 = { 0, 0, 0, 0, 0, 0, 0, 0 };
    __builtin_amdgcn_tensor_load_to_lds(g0, g1, gz, gz, gz8, 0);
#else
    __builtin_amdgcn_tensor_load_to_lds(g0, g1, gz, gz, 0);
#endif
}
#endif

// ---------------------------------------------------------------------------
// Kernel 1: QKV = bf16( X[16384,512] @ W[512,1536] + bias )
// Tile: 256(M) x 64(N); each wave owns 32 rows x 64 cols (8 WMMA / K-chunk).
// ---------------------------------------------------------------------------
__global__ __launch_bounds__(256)
void qkv_gemm_kernel(const float* __restrict__ X,
                     const float* __restrict__ W,
                     const float* __restrict__ bias,
                     unsigned short* __restrict__ QKV) {
    __shared__ alignas(16) unsigned short Abf[256 * 32];   // [row][kk] 16KB
    __shared__ alignas(16) unsigned short Btl[64 * 32];    // [n][kk]   4KB

    const int n0   = blockIdx.x * 64;
    const int m0   = blockIdx.y * 256;
    const int t    = threadIdx.x;
    const int wave = t >> 5;
    const int lane = t & 31;
    const int hl   = (lane >= 16) ? 1 : 0;
    const int ln   = lane & 15;

    v8f acc[2][4];
#pragma unroll
    for (int g = 0; g < 2; ++g)
#pragma unroll
        for (int s = 0; s < 4; ++s)
#pragma unroll
            for (int j = 0; j < 8; ++j) acc[g][s][j] = 0.0f;

#pragma unroll 1
    for (int k0 = 0; k0 < D_HID; k0 += 32) {
        __syncthreads();
        // stage A: 256 rows x 32 k (fp32 -> bf16)
#pragma unroll
        for (int i = 0; i < 8; ++i) {
            int f   = t + 256 * i;          // 2048 float4s
            int row = f >> 3;               // 8 float4 per row
            int c4  = (f & 7) * 4;
            fv4 v = *(const fv4*)(X + (size_t)(m0 + row) * D_HID + k0 + c4);
            unsigned short* dst = Abf + row * 32 + c4;
            dst[0] = f2bf(v[0]); dst[1] = f2bf(v[1]);
            dst[2] = f2bf(v[2]); dst[3] = f2bf(v[3]);
        }
        // stage B transposed: Btl[n][kk] = W[k0+kk][n0+n]
#pragma unroll
        for (int i = 0; i < 2; ++i) {
            int f  = t + 256 * i;           // 512 float4s
            int kk = f >> 4;                // 16 float4 per kk row
            int n4 = (f & 15) * 4;
            fv4 v = *(const fv4*)(W + (size_t)(k0 + kk) * N_QKV + n0 + n4);
            Btl[(n4 + 0) * 32 + kk] = f2bf(v[0]);
            Btl[(n4 + 1) * 32 + kk] = f2bf(v[1]);
            Btl[(n4 + 2) * 32 + kk] = f2bf(v[2]);
            Btl[(n4 + 3) * 32 + kk] = f2bf(v[3]);
        }
        __syncthreads();

        Frag a[2];
#pragma unroll
        for (int g = 0; g < 2; ++g) {
            const uv4* arow = (const uv4*)(Abf + (wave * 32 + g * 16 + ln) * 32);
            a[g].q[0] = arow[hl * 2];       // kk 0..7  | 16..23
            a[g].q[1] = arow[hl * 2 + 1];   // kk 8..15 | 24..31
        }
#pragma unroll
        for (int s = 0; s < 4; ++s) {
            Frag b;
            const uv4* brow = (const uv4*)(Btl + (s * 16 + ln) * 32);
            b.q[0] = brow[hl * 2];          // kk 0..7  | 16..23
            b.q[1] = brow[hl * 2 + 1];      // kk 8..15 | 24..31
            acc[0][s] = wmma_bf16<false, false>(a[0], b, acc[0][s]);
            acc[1][s] = wmma_bf16<false, true >(a[1], b, acc[1][s]); // reuse B
        }
    }

    // epilogue: add bias, convert to bf16, store
#pragma unroll
    for (int g = 0; g < 2; ++g)
#pragma unroll
        for (int s = 0; s < 4; ++s) {
            int col  = n0 + s * 16 + ln;
            float bv = bias[col];
#pragma unroll
            for (int r = 0; r < 8; ++r) {
                int row = m0 + wave * 32 + g * 16 + r + hl * 8;
                QKV[(size_t)row * N_QKV + col] = f2bf(acc[g][s][r] + bv);
            }
        }
}

// ---------------------------------------------------------------------------
// Kernel 2: flash attention. 1 workgroup = 128 query rows (8 waves x 16).
// Pipelined TDM: while waves stage V^T and run the PV WMMAs, the Tensor Data
// Mover is already pulling the NEXT K tile into its own LDS buffer.
// ---------------------------------------------------------------------------
__global__ __launch_bounds__(256)
void flash_attn_kernel(const unsigned short* __restrict__ QKV,
                       const int* __restrict__ amask,
                       float* __restrict__ out) {
    __shared__ alignas(16) unsigned short Kbuf[32 * 512];    // K tile     32KB
    __shared__ alignas(16) unsigned short Vt[512 * 32];      // V^T tile   32KB
    __shared__ alignas(16) unsigned short Pl[8 * 16 * 32];   // P patches   8KB

    const int b    = blockIdx.x >> 5;          // batch
    const int q0   = (blockIdx.x & 31) * 128;  // query block base
    const int t    = threadIdx.x;
    const int wave = t >> 5;
    const int lane = t & 31;
    const int hl   = (lane >= 16) ? 1 : 0;
    const int ln   = lane & 15;

    const unsigned short* Qg = QKV + (size_t)b * S_LEN * N_QKV;
    const unsigned short* Kg = Qg + 512;
    const unsigned short* Vg = Qg + 1024;

    // preload this wave's 16x512 Q panel as 16 A-fragments
    Frag qf[16];
    {
        const size_t qbase = (size_t)(q0 + wave * 16 + ln) * N_QKV;
#pragma unroll
        for (int c = 0; c < 16; ++c) {
            const unsigned short* p = Qg + qbase + c * 32 + hl * 16;
            qf[c].q[0] = *(const uv4*)(p);
            qf[c].q[1] = *(const uv4*)(p + 8);
        }
    }

    float m[8], l[8];
    v8f   o[32];
#pragma unroll
    for (int r = 0; r < 8; ++r) { m[r] = -3.0e38f; l[r] = 0.0f; }
#pragma unroll
    for (int s = 0; s < 32; ++s)
#pragma unroll
        for (int j = 0; j < 8; ++j) o[s][j] = 0.0f;

    unsigned short* pw = Pl + wave * 512;      // wave-private 16x32 P patch

#ifdef USE_TDM
    if (wave == 0) tdm_load_tile(Kg, Kbuf);    // prefetch K tile 0
#endif

#pragma unroll 1
    for (int kt = 0; kt < S_LEN / 32; ++kt) {
        const int key0 = kt * 32;

#ifdef USE_TDM
        if (wave == 0) __builtin_amdgcn_s_wait_tensorcnt(0);
#else
        // cooperative copy, Kbuf[key*512 + dim]
#pragma unroll
        for (int i = 0; i < 8; ++i) {
            int f   = t + 256 * i;             // 2048 x 16B
            int key = f >> 6;
            int g   = (f & 63) * 8;
            *(uv4*)(Kbuf + key * 512 + g) =
                *(const uv4*)(Kg + (size_t)(key0 + key) * N_QKV + g);
        }
#endif
        __syncthreads();                       // K tile visible to all waves

        // ---- scores: S = Q @ K^T (2 key subtiles x 16 k-chunks) ----
        v8f sc[2];
#pragma unroll
        for (int s = 0; s < 2; ++s)
#pragma unroll
            for (int j = 0; j < 8; ++j) sc[s][j] = 0.0f;
#pragma unroll
        for (int c = 0; c < 16; ++c) {
            Frag kb0, kb1;
            const unsigned short* kr0 = Kbuf + ln * 512 + c * 32 + hl * 16;
            const unsigned short* kr1 = kr0 + 16 * 512;
            kb0.q[0] = *(const uv4*)(kr0);
            kb0.q[1] = *(const uv4*)(kr0 + 8);
            kb1.q[0] = *(const uv4*)(kr1);
            kb1.q[1] = *(const uv4*)(kr1 + 8);
            sc[0] = wmma_bf16<false, false>(qf[c], kb0, sc[0]);
            sc[1] = wmma_bf16<true,  false>(qf[c], kb1, sc[1]);  // reuse A
        }

        // ---- mask (before scaling, like reference) + online softmax ----
        const int mk0 = amask[b * S_LEN + key0 + ln];
        const int mk1 = amask[b * S_LEN + key0 + 16 + ln];
#pragma unroll
        for (int r = 0; r < 8; ++r) {
            if (mk0 == 0) sc[0][r] = MASKFILL;
            if (mk1 == 0) sc[1][r] = MASKFILL;
        }
        float mt[8];
#pragma unroll
        for (int r = 0; r < 8; ++r) mt[r] = fmaxf(sc[0][r], sc[1][r]);
#pragma unroll
        for (int r = 0; r < 8; ++r)
#pragma unroll
            for (int off = 1; off < 16; off <<= 1)
                mt[r] = fmaxf(mt[r], __shfl_xor(mt[r], off, 32));

        float corr[8];
#pragma unroll
        for (int r = 0; r < 8; ++r) {
            float mn = fmaxf(m[r], mt[r]);
            corr[r]  = __expf((m[r] - mn) * SCALE_F);
            m[r]     = mn;
            sc[0][r] = __expf((sc[0][r] - mn) * SCALE_F);
            sc[1][r] = __expf((sc[1][r] - mn) * SCALE_F);
        }
#pragma unroll
        for (int r = 0; r < 8; ++r) {
            float ssum = sc[0][r] + sc[1][r];
#pragma unroll
            for (int off = 1; off < 16; off <<= 1)
                ssum += __shfl_xor(ssum, off, 32);
            l[r] = l[r] * corr[r] + ssum;
        }
#pragma unroll
        for (int s = 0; s < 32; ++s)
#pragma unroll
            for (int r = 0; r < 8; ++r) o[s][r] = o[s][r] * corr[r];

        // write P (bf16) to wave-private LDS patch: pw[row*32 + key]
#pragma unroll
        for (int s = 0; s < 2; ++s)
#pragma unroll
            for (int r = 0; r < 8; ++r)
                pw[(r + hl * 8) * 32 + s * 16 + ln] = f2bf(sc[s][r]);

        __syncthreads();                       // all waves done reading Kbuf

#ifdef USE_TDM
        // pipeline: DMA of the NEXT K tile overlaps V staging + PV WMMAs
        if (kt + 1 < S_LEN / 32 && wave == 0)
            tdm_load_tile(Kg + (size_t)(key0 + 32) * N_QKV, Kbuf);
#endif

        // stage V transposed: Vt[dim*32 + key]
#pragma unroll
        for (int i = 0; i < 8; ++i) {
            int f   = t + 256 * i;
            int key = f >> 6;
            int g   = (f & 63) * 8;
            U4H v;
            v.q = *(const uv4*)(Vg + (size_t)(key0 + key) * N_QKV + g);
#pragma unroll
            for (int j = 0; j < 8; ++j) Vt[(g + j) * 32 + key] = v.h[j];
        }
        __syncthreads();                       // V^T + P visible

        // ---- PV: O += P @ V (1 chunk of K=32, 32 dim subtiles) ----
        Frag pa;
        {
            const unsigned short* pr = pw + ln * 32 + hl * 16;
            pa.q[0] = *(const uv4*)(pr);
            pa.q[1] = *(const uv4*)(pr + 8);
        }
        {   // s = 0 peeled (no reuse), then constant reuse_a chain
            Frag vb;
            const unsigned short* vr = Vt + ln * 32 + hl * 16;
            vb.q[0] = *(const uv4*)(vr);
            vb.q[1] = *(const uv4*)(vr + 8);
            o[0] = wmma_bf16<false, false>(pa, vb, o[0]);
        }
#pragma unroll
        for (int s = 1; s < 32; ++s) {
            Frag vb;
            const unsigned short* vr = Vt + (s * 16 + ln) * 32 + hl * 16;
            vb.q[0] = *(const uv4*)(vr);
            vb.q[1] = *(const uv4*)(vr + 8);
            o[s] = wmma_bf16<true, false>(pa, vb, o[s]);           // reuse A
        }
    }

    // ---- epilogue: normalize and store fp32 ----
#pragma unroll
    for (int r = 0; r < 8; ++r) l[r] = 1.0f / l[r];
    const int row_base = b * S_LEN + q0 + wave * 16;
#pragma unroll
    for (int s = 0; s < 32; ++s)
#pragma unroll
        for (int r = 0; r < 8; ++r) {
            int row = row_base + r + hl * 8;
            out[(size_t)row * D_HID + s * 16 + ln] = o[s][r] * l[r];
        }
}

// ---------------------------------------------------------------------------
extern "C" void kernel_launch(void* const* d_in, const int* in_sizes, int n_in,
                              void* d_out, int out_size, void* d_ws, size_t ws_size,
                              hipStream_t stream) {
    (void)in_sizes; (void)n_in; (void)out_size; (void)ws_size;
    const float* X    = (const float*)d_in[0];
    const float* W    = (const float*)d_in[1];
    const float* bias = (const float*)d_in[2];
    const int*   msk  = (const int*)d_in[3];
    float*       out  = (float*)d_out;

    // workspace: QKV in bf16, 16384 x 1536 x 2B = 48 MB
    unsigned short* QKV = (unsigned short*)d_ws;

    dim3 g1(N_QKV / 64, (4 * S_LEN) / 256);              // 24 x 64
    qkv_gemm_kernel<<<g1, 256, 0, stream>>>(X, W, bias, QKV);

    dim3 g2((4 * S_LEN) / 128);                          // 128 workgroups
    flash_attn_kernel<<<g2, 256, 0, stream>>>(QKV, msk, out);
}
